// frameRNN_dec_34308198760908
// MI455X (gfx1250) — matrix operand order repaired and checked by reading, not compiled, not run on hardware
//
#include <hip/hip_runtime.h>
#include <hip/hip_bf16.h>
#include <math.h>

// ---------------------------------------------------------------------------
// Problem constants (from reference setup_inputs)
// ---------------------------------------------------------------------------
constexpr int    Bn  = 8;
constexpr int    Tn  = 4000;
constexpr int    Cn  = 4;
constexpr int    Dn  = 256;
constexpr int    Gn  = 768;           // 3*Dn (GRU gates)
constexpr size_t BTn = (size_t)Bn * Tn;   // 32000 rows (multiple of 16)

typedef __attribute__((ext_vector_type(16))) _Float16 v16h;
typedef __attribute__((ext_vector_type(8)))  float    v8f;

// ---------------------------------------------------------------------------
// Workspace layout (bytes)
// ---------------------------------------------------------------------------
constexpr size_t AL(size_t x) { return (x + 255) & ~size_t(255); }
constexpr size_t OFF_ENC_H  = 0;
constexpr size_t OFF_WP_H   = AL(OFF_ENC_H  + 2*BTn*Dn);
constexpr size_t OFF_WIA_H  = AL(OFF_WP_H   + 2*(size_t)Dn*Dn);
constexpr size_t OFF_WHA_H  = AL(OFF_WIA_H  + 2*(size_t)Gn*Dn);
constexpr size_t OFF_WIM_H  = AL(OFF_WHA_H  + 2*(size_t)Gn*Dn);
constexpr size_t OFF_WHM_H  = AL(OFF_WIM_H  + 2*(size_t)Gn*Dn);
constexpr size_t OFF_FEMB   = AL(OFF_WHM_H  + 2*(size_t)Gn*Dn);   // f32, mutated per speaker
constexpr size_t OFF_FEMB_H = AL(OFF_FEMB   + 4*BTn*Dn);          // f16 copy for GEMM A
constexpr size_t OFF_SEQ_H  = AL(OFF_FEMB_H + 2*BTn*Dn);          // gathered seq, f16
constexpr size_t OFF_XP     = AL(OFF_SEQ_H  + 2*BTn*Dn);          // x_pre f16 (reused a/m)
constexpr size_t OFF_OUT    = AL(OFF_XP     + 2*BTn*Gn);          // GRU scan outputs
constexpr size_t OFF_UPD    = AL(OFF_OUT    + 4*BTn*Dn);          // memory-GRU update
constexpr size_t OFF_HPM    = AL(OFF_UPD    + 4*BTn*Dn);          // 8x768
constexpr size_t OFF_ADDH   = AL(OFF_HPM    + 4*(size_t)Bn*Gn);   // 8x256
constexpr size_t OFF_PERM   = AL(OFF_ADDH   + 4*(size_t)Bn*Dn);
constexpr size_t OFF_MASK   = AL(OFF_PERM   + 4*BTn);
constexpr size_t OFF_ACT    = AL(OFF_MASK   + 4*BTn);
constexpr size_t OFF_CEXCL  = AL(OFF_ACT    + 4*BTn);
constexpr size_t OFF_SIF    = AL(OFF_CEXCL  + 4*BTn);             // spk_in_frame
constexpr size_t OFF_ORDER  = AL(OFF_SIF    + 4*BTn);
constexpr size_t OFF_ACC    = AL(OFF_ORDER  + 4*(size_t)Bn*Cn);   // [0:8)=spk, [8:16)=fal, [16:24)=valid

// ---------------------------------------------------------------------------
// Device helpers
// ---------------------------------------------------------------------------
__device__ inline float sigmoidf_(float x) { return 1.f / (1.f + expf(-x)); }
__device__ inline float bce_(float p, float y) {
  float lp = fmaxf(logf(p),     -100.f);
  float lq = fmaxf(log1pf(-p),  -100.f);
  return -(y * lp + (1.f - y) * lq);
}
__device__ inline float waveReduce(float v) {
  #pragma unroll
  for (int o = 16; o > 0; o >>= 1) v += __shfl_down(v, o, 32);
  return v;
}
// 256-thread block reduce with broadcast; sbuf must hold >=8 floats
__device__ inline float blockReduceB(float v, float* sbuf) {
  int lane = threadIdx.x & 31, w = threadIdx.x >> 5;
  v = waveReduce(v);
  __syncthreads();
  if (lane == 0) sbuf[w] = v;
  __syncthreads();
  if (threadIdx.x == 0) {
    float s = 0.f;
    int nw = blockDim.x >> 5;
    for (int i = 0; i < nw; ++i) s += sbuf[i];
    sbuf[0] = s;
  }
  __syncthreads();
  float r = sbuf[0];
  __syncthreads();
  return r;
}

// WMMA wrapper: D = A(16x32 f16) * B(32x16 f16) + C (f32)
__device__ inline v8f wmma16x16x32(v16h a, v16h b, v8f c) {
  return __builtin_amdgcn_wmma_f32_16x16x32_f16(
      /*neg_a=*/false, a, /*neg_b=*/false, b,
      /*c_mod=*/(short)0, c, /*reuse_a=*/false, /*reuse_b=*/false);
}

// CDNA5 async DMA: copy 16B from global to LDS, tracked by ASYNCcnt.
// lds_off = byte offset within the workgroup's LDS allocation (addr[31:0]
// of the generic pointer per the flat-aperture mapping).
__device__ inline void async_load_b128(unsigned lds_off, const void* gptr) {
  asm volatile("global_load_async_to_lds_b128 %0, %1, off"
               :: "v"(lds_off), "v"(gptr) : "memory");
}

// A fragment: 16x32 tile at (row0,k0) of row-major [M x lda] f16 matrix.
// Layout: lanes 0-15 rows M=lane, K=0..15; lanes 16-31 K=16..31 (matches load_b_frag).
__device__ inline v16h load_a_frag(const _Float16* __restrict__ A, int lda,
                                   int row0, int k0, int lane) {
  int m  = row0 + (lane & 15);
  int hi = (lane >> 4) & 1;
  v16h a;
  #pragma unroll
  for (int e = 0; e < 16; ++e) {
    int k = k0 + e + hi * 16;
    a[e] = A[(size_t)m * lda + k];
  }
  return a;
}
// B fragment for D = A * W^T: logical B[k][n] = W[n][k]; W row-major [N x ldw].
// 32x16 tile at (k0, n0). lanes 0-15: col N=lane, K=0..15; lanes 16-31: K=16..31.
__device__ inline v16h load_b_frag(const _Float16* __restrict__ W, int ldw,
                                   int n0, int k0, int lane) {
  int n  = n0 + (lane & 15);
  int hi = (lane >> 4) & 1;
  v16h b;
  #pragma unroll
  for (int e = 0; e < 16; ++e) {
    int k = k0 + e + hi * 16;
    b[e] = W[(size_t)n * ldw + k];
  }
  return b;
}

// ---------------------------------------------------------------------------
// Kernels
// ---------------------------------------------------------------------------
__global__ void k_init(float* acc) {
  if (threadIdx.x < 24) acc[threadIdx.x] = 0.f;
}

__global__ void k_f32_to_f16(const float* __restrict__ src, _Float16* __restrict__ dst, int n) {
  int stride = gridDim.x * blockDim.x;
  for (int i = blockIdx.x * blockDim.x + threadIdx.x; i < n; i += stride)
    dst[i] = (_Float16)src[i];
}

// Generic  C = A(f16, [M x 256]) * W(f16,[N x 256])^T + bias ; f32 and/or f16 out.
// One 16x16 output tile per wave; 8 waves / 256-thread block; K = 256 unrolled (8 WMMAs).
__global__ void k_gemm_f16(const _Float16* __restrict__ A, const _Float16* __restrict__ W,
                           const float* __restrict__ bias, float* __restrict__ Cf,
                           _Float16* __restrict__ Ch, int M, int N) {
  int wid    = blockIdx.x * 8 + (threadIdx.x >> 5);
  int ntiles = N >> 4;
  int total  = (M >> 4) * ntiles;
  if (wid >= total) return;
  int mt = wid / ntiles, nt = wid % ntiles;
  int lane = threadIdx.x & 31;
  v8f acc = {};
  #pragma unroll
  for (int kk = 0; kk < 8; ++kk) {
    // WGP-scope prefetch of the next K-slice of A (global_prefetch_b8)
    __builtin_prefetch(A + (size_t)(mt * 16 + (lane & 15)) * 256 + kk * 32 + 32, 0, 3);
    v16h a = load_a_frag(A, 256, mt * 16, kk * 32, lane);
    v16h b = load_b_frag(W, 256, nt * 16, kk * 32, lane);
    acc = wmma16x16x32(a, b, acc);
  }
  int   hi  = (lane >> 4) & 1;
  int   col = nt * 16 + (lane & 15);
  float bv  = bias ? bias[col] : 0.f;
  #pragma unroll
  for (int r = 0; r < 8; ++r) {
    int    row = mt * 16 + r + hi * 8;
    size_t idx = (size_t)row * N + col;
    float  v   = acc[r] + bv;
    if (Cf) Cf[idx] = v;
    if (Ch) Ch[idx] = (_Float16)v;
  }
}

// Per-batch VAD statistics + base frame_active_loss; also builds spk_in_frame.
__global__ void k_vad(const int* __restrict__ label, const int* __restrict__ seq_len,
                      const float* __restrict__ femb, const float* __restrict__ h0,
                      float* __restrict__ sif, float* __restrict__ acc) {
  int b = blockIdx.x, tid = threadIdx.x;
  __shared__ float sbuf[32];
  int   L  = seq_len[b];
  float Tb = (float)L;
  float lpos = 0.f, lsif = 0.f;
  for (int t = tid; t < Tn; t += 256) {
    float s = 0.f;
    #pragma unroll
    for (int c = 0; c < Cn; ++c) s += (float)label[((size_t)(b * Tn + t)) * Cn + c];
    sif[b * Tn + t] = s;
    float vl = (s > 0.f) ? 1.f : 0.f;
    if (t < L) lpos += vl;
    lsif += s;
  }
  float pos    = blockReduceB(lpos, sbuf);
  float sumsif = blockReduceB(lsif, sbuf);
  if (tid == 0) acc[16 + b] = Tb + sumsif;     // valid_frames
  int   lane = tid & 31, wave = tid >> 5;
  float lacc = 0.f;
  for (int t = wave; t < Tn; t += 8) {
    float d = 0.f;
    #pragma unroll
    for (int j = 0; j < 8; ++j) {
      int k = lane + 32 * j;
      d += femb[((size_t)(b * Tn + t)) * Dn + k] * h0[k];
    }
    d = waveReduce(d);
    if (lane == 0 && t < L) {
      float p  = sigmoidf_(d);
      float vl = (sif[b * Tn + t] > 0.f) ? 1.f : 0.f;
      float w  = (vl == 1.f) ? Tb / (pos + 1e-9f) : Tb / (Tb - pos + 1e-9f);
      lacc += w * bce_(p, vl);
    }
  }
  __syncthreads();
  if (lane == 0) sbuf[wave] = lacc;
  __syncthreads();
  if (tid == 0) {
    float s = 0.f;
    for (int w = 0; w < 8; ++w) s += sbuf[w];
    acc[8 + b] += s;
  }
}

// first_active per (b,c) + stable argsort over C=4 speakers.
__global__ void k_order(const int* __restrict__ label, int* __restrict__ order) {
  __shared__ int sfa[32];
  int tid = threadIdx.x;
  if (tid < 32) {
    int b = tid >> 2, c = tid & 3;
    int fa = Tn + 1;
    for (int t = 0; t < Tn; ++t) {
      if (label[((size_t)(b * Tn + t)) * Cn + c] != 0) { fa = t; break; }
    }
    sfa[tid] = fa;
  }
  __syncthreads();
  if (tid < Bn) {
    int b = tid;
    for (int c = 0; c < Cn; ++c) {
      int fc = sfa[b * 4 + c];
      int rank = 0;
      for (int c2 = 0; c2 < Cn; ++c2) {
        int f2 = sfa[b * 4 + c2];
        if (f2 < fc || (f2 == fc && c2 < c)) rank++;
      }
      order[b * 4 + rank] = c;
    }
  }
}

// Stable partition (actives first, time order) -> perm/mask; act + exclusive cumsum.
__global__ void k_perm(const int* __restrict__ label, const int* __restrict__ order, int K,
                       int* __restrict__ perm, int* __restrict__ mask,
                       int* __restrict__ act, int* __restrict__ cexcl) {
  if (threadIdx.x != 0) return;
  int b   = blockIdx.x;
  int spk = order[b * 4 + K];
  int na  = 0;
  for (int t = 0; t < Tn; ++t) na += (label[((size_t)(b * Tn + t)) * Cn + spk] != 0);
  int ia = 0, ii = 0, cum = 0;
  for (int t = 0; t < Tn; ++t) {
    int a = (label[((size_t)(b * Tn + t)) * Cn + spk] != 0);
    act[b * Tn + t]   = a;
    cexcl[b * Tn + t] = cum;
    cum += a;
    int pos = a ? (ia++) : (na + ii++);
    perm[b * Tn + pos] = t;
    mask[b * Tn + pos] = a;
  }
}

__global__ void k_gather_seq(const float* __restrict__ femb, const int* __restrict__ perm,
                             _Float16* __restrict__ seq_h) {
  size_t total = BTn * Dn;
  for (size_t idx = (size_t)blockIdx.x * 256 + threadIdx.x; idx < total;
       idx += (size_t)gridDim.x * 256) {
    int bi = (int)(idx >> 8);
    int d  = (int)(idx & 255);
    int b  = bi / Tn;
    int t  = perm[bi];
    seq_h[idx] = (_Float16)femb[((size_t)(b * Tn + t)) * Dn + d];
  }
}

// Serial GRU scan: 1 block, 512 threads = 16 waves x 3 N-tiles = 48 tiles of the
// 16x768 gate matmul (batch 8 padded to M=16). Wh fragments live in VGPRs; the
// per-step x_pre slab (8x768 f16 = 12KB) is double-buffered into LDS with CDNA5
// async DMA (global_load_async_to_lds_b128) overlapped with the WMMA phase.
__global__ void __launch_bounds__(512)
k_gru_scan(const _Float16* __restrict__ Wh_h, const float* __restrict__ bh,
           const _Float16* __restrict__ xph, const int* __restrict__ mask,
           const float* __restrict__ h0, float* __restrict__ out,
           float* __restrict__ adder_h) {
  __shared__ _Float16 sA[16 * 256];                    // padded h in f16 (rows 8..15 zero)
  __shared__ float    sh[8 * 256];                     // h (f32)
  __shared__ _Float16 shp[8 * 768];                    // h @ Wh^T (f16)
  __shared__ __align__(16) _Float16 sxp[2][8 * 768];   // x_pre ping-pong
  int tid = threadIdx.x, lane = tid & 31, wave = tid >> 5;

  for (int i = tid; i < 8 * 256; i += 512)  sh[i] = h0[i & 255];
  for (int i = tid; i < 16 * 256; i += 512) sA[i] = (_Float16)0.f;

  // Kick off async prefetch of step 0's x_pre slab into sxp[0].
  // 8x768 f16 = 768 x 16B chunks; threads 0-255 issue 3 chunks each -> every
  // wave that issues has exactly 3 ASYNCcnt increments per step.
  if (tid < 256) {
    unsigned base0 = (unsigned)(size_t)(&sxp[0][0]);
    #pragma unroll
    for (int j = 0; j < 3; ++j) {
      int i = tid + j * 256;                 // chunk id 0..767
      int b = i / 96, ch = i % 96;           // 96 chunks per 768-elem row
      const _Float16* g = xph + ((size_t)(b * Tn + 0)) * Gn + ch * 8;
      async_load_b128(base0 + (unsigned)i * 16u, g);
    }
  }
  __syncthreads();

  // Preload this wave's 3 B-tiles x 8 K-slices of Wh (invariant across 4000 steps)
  v16h bw[3][8];
  #pragma unroll
  for (int j = 0; j < 3; ++j) {
    int nt = wave * 3 + j;
    #pragma unroll
    for (int kk = 0; kk < 8; ++kk)
      bw[j][kk] = load_b_frag(Wh_h, 256, nt * 16, kk * 32, lane);
  }

  for (int t = 0; t < Tn; ++t) {
    int buf = t & 1, nbuf = buf ^ 1;
    for (int i = tid; i < 8 * 256; i += 512) sA[i] = (_Float16)sh[i];
    __syncthreads();

    // Issue async DMA for step t+1 (clamped; uniform 3 issues/wave on waves 0-7)
    {
      int tnext = (t + 1 < Tn) ? (t + 1) : (Tn - 1);
      if (tid < 256) {
        unsigned baseN = (unsigned)(size_t)(&sxp[nbuf][0]);
        #pragma unroll
        for (int j = 0; j < 3; ++j) {
          int i = tid + j * 256;
          int b = i / 96, ch = i % 96;
          const _Float16* g = xph + ((size_t)(b * Tn + tnext)) * Gn + ch * 8;
          async_load_b128(baseN + (unsigned)i * 16u, g);
        }
      }
    }

    // WMMA phase (overlaps the in-flight DMA)
    v8f acc0 = {}, acc1 = {}, acc2 = {};
    #pragma unroll
    for (int kk = 0; kk < 8; ++kk) {
      v16h a = load_a_frag(sA, 256, 0, kk * 32, lane);   // LDS -> ds_load
      acc0 = wmma16x16x32(a, bw[0][kk], acc0);
      acc1 = wmma16x16x32(a, bw[1][kk], acc1);
      acc2 = wmma16x16x32(a, bw[2][kk], acc2);
    }
    int hi = (lane >> 4) & 1;
    if (hi == 0) {                                        // lanes 0-15 hold rows 0-7
      int col = lane & 15;
      #pragma unroll
      for (int r = 0; r < 8; ++r) {
        shp[r * 768 + (wave * 3 + 0) * 16 + col] = (_Float16)acc0[r];
        shp[r * 768 + (wave * 3 + 1) * 16 + col] = (_Float16)acc1[r];
        shp[r * 768 + (wave * 3 + 2) * 16 + col] = (_Float16)acc2[r];
      }
    }
    // Step t's 3 async issues are the oldest in flight; allow t+1's 3 to remain.
    asm volatile("s_wait_asynccnt 3" ::: "memory");
    __syncthreads();

    const _Float16* xrow = sxp[buf];
    for (int i = tid; i < 8 * 256; i += 512) {
      int b = i >> 8, d = i & 255;
      float xr = (float)xrow[b * 768 + d];
      float xz = (float)xrow[b * 768 + 256 + d];
      float xn = (float)xrow[b * 768 + 512 + d];
      float hr = (float)shp[b * 768 + d]       + bh[d];
      float hz = (float)shp[b * 768 + 256 + d] + bh[256 + d];
      float hn = (float)shp[b * 768 + 512 + d] + bh[512 + d];
      float r = sigmoidf_(xr + hr);
      float z = sigmoidf_(xz + hz);
      float n = tanhf(xn + r * hn);
      float hprev = sh[i];
      float hnew  = (1.f - z) * n + z * hprev;
      out[(size_t)(b * Tn + t) * Dn + d] = hnew;
      sh[i] = mask[b * Tn + t] ? hnew : hprev;
    }
    __syncthreads();
  }
  for (int i = tid; i < 8 * 256; i += 512) adder_h[i] = sh[i];
}

// Speaker attractor loss (score/BCE over valid frames) -> acc[0:8)
__global__ void k_score(const float* __restrict__ femb, const float* __restrict__ out,
                        const int* __restrict__ act, const int* __restrict__ cexcl,
                        const int* __restrict__ seq_len, float* __restrict__ acc) {
  int b = blockIdx.x, tid = threadIdx.x;
  __shared__ float sbuf[32];
  int L = seq_len[b];
  float lN = 0.f, lP = 0.f;
  for (int t = tid; t < Tn; t += 256) {
    bool valid = (cexcl[b * Tn + t] >= 1) && (t < L);
    if (valid) { lN += 1.f; if (act[b * Tn + t]) lP += 1.f; }
  }
  float Nf   = blockReduceB(lN, sbuf);
  float posN = blockReduceB(lP, sbuf);
  int   lane = tid & 31, wave = tid >> 5;
  float lacc = 0.f;
  for (int t = wave; t < Tn; t += 8) {
    int  ce    = cexcl[b * Tn + t];
    bool valid = (ce >= 1) && (t < L);
    if (!valid) continue;                       // uniform per wave (t is wave-uniform)
    int jj = ce - 1; if (jj < 0) jj = 0; if (jj > Tn - 1) jj = Tn - 1;
    float d = 0.f;
    #pragma unroll
    for (int j = 0; j < 8; ++j) {
      int k = lane + 32 * j;
      d += femb[((size_t)(b * Tn + t)) * Dn + k] * out[((size_t)(b * Tn + jj)) * Dn + k];
    }
    d = waveReduce(d);
    if (lane == 0) {
      float p  = sigmoidf_(d);
      float sl = act[b * Tn + t] ? 1.f : 0.f;
      float w2 = (sl == 1.f) ? Nf / (posN + 1e-9f) : Nf / (Nf - posN + 1e-9f);
      lacc += w2 * bce_(p, sl);
    }
  }
  __syncthreads();
  if (lane == 0) sbuf[wave] = lacc;
  __syncthreads();
  if (tid == 0) {
    float s = 0.f;
    for (int w = 0; w < 8; ++w) s += sbuf[w];
    acc[b] += (Nf > 0.f) ? s / fmaxf(Nf, 1.f) : 0.f;
  }
}

// hp_m = adder_h @ Wh_m^T + bh_m  (8x768 via one padded 16x768 WMMA pass)
__global__ void __launch_bounds__(512)
k_hp_m(const float* __restrict__ adder_h, const _Float16* __restrict__ Whm_h,
       const float* __restrict__ bh_m, float* __restrict__ hpm) {
  __shared__ _Float16 sA[16 * 256];
  int tid = threadIdx.x, lane = tid & 31, wave = tid >> 5;
  for (int i = tid; i < 16 * 256; i += 512)
    sA[i] = (_Float16)((i < 8 * 256) ? adder_h[i] : 0.f);
  __syncthreads();
  #pragma unroll
  for (int j = 0; j < 3; ++j) {
    int nt = wave * 3 + j;
    v8f acc = {};
    #pragma unroll
    for (int kk = 0; kk < 8; ++kk) {
      v16h a = load_a_frag(sA, 256, 0, kk * 32, lane);
      v16h bf = load_b_frag(Whm_h, 256, nt * 16, kk * 32, lane);
      acc = wmma16x16x32(a, bf, acc);
    }
    if (((lane >> 4) & 1) == 0) {
      int col = nt * 16 + (lane & 15);
      #pragma unroll
      for (int r = 0; r < 8; ++r) hpm[r * 768 + col] = acc[r] + bh_m[col];
    }
  }
}

// Memory-GRU one step elementwise: upd = (1-z)*n + z*adder_h
__global__ void k_upd(const _Float16* __restrict__ xpm, const float* __restrict__ hpm,
                      const float* __restrict__ adder_h, float* __restrict__ upd) {
  size_t total = BTn * Dn;
  for (size_t idx = (size_t)blockIdx.x * 256 + threadIdx.x; idx < total;
       idx += (size_t)gridDim.x * 256) {
    int    bi = (int)(idx >> 8);
    int    d  = (int)(idx & 255);
    int    b  = bi / Tn;
    size_t base = (size_t)bi * Gn;
    float xr = (float)xpm[base + d];
    float xz = (float)xpm[base + 256 + d];
    float xn = (float)xpm[base + 512 + d];
    float hr = hpm[b * 768 + d], hz = hpm[b * 768 + 256 + d], hn = hpm[b * 768 + 512 + d];
    float r = sigmoidf_(xr + hr);
    float z = sigmoidf_(xz + hz);
    float n = tanhf(xn + r * hn);
    upd[idx] = (1.f - z) * n + z * adder_h[b * 256 + d];
  }
}

// Scatter updated embeddings, update spk_in_frame, add residual-VAD loss -> acc[8:16)
__global__ void k_scatter(const float* __restrict__ upd, const int* __restrict__ perm,
                          const int* __restrict__ mask, const int* __restrict__ act,
                          const float* __restrict__ h0, float* __restrict__ sif,
                          float* __restrict__ femb, _Float16* __restrict__ femb_h,
                          float* __restrict__ acc) {
  int b = blockIdx.x, tid = threadIdx.x;
  __shared__ float sbuf[32];
  for (int t = tid; t < Tn; t += 256) sif[b * Tn + t] -= (float)act[b * Tn + t];
  __syncthreads();
  float lL = 0.f, lp3 = 0.f;
  for (int i = tid; i < Tn; i += 256) {
    if (mask[b * Tn + i]) {
      lL += 1.f;
      int t = perm[b * Tn + i];
      if (sif[b * Tn + t] > 0.f) lp3 += 1.f;
    }
  }
  float Lb   = blockReduceB(lL, sbuf);
  float pos3 = blockReduceB(lp3, sbuf);
  int   lane = tid & 31, wave = tid >> 5;
  float lacc = 0.f;
  for (int i = wave; i < Tn; i += 8) {
    if (!mask[b * Tn + i]) continue;            // wave-uniform
    int   t = perm[b * Tn + i];
    float d = 0.f;
    #pragma unroll
    for (int j = 0; j < 8; ++j) {
      int   k = lane + 32 * j;
      float u = upd[((size_t)(b * Tn + i)) * Dn + k];
      d += u * h0[k];
      femb  [((size_t)(b * Tn + t)) * Dn + k] = u;
      femb_h[((size_t)(b * Tn + t)) * Dn + k] = (_Float16)u;
    }
    d = waveReduce(d);
    if (lane == 0) {
      float p  = sigmoidf_(d);
      float l3 = (sif[b * Tn + t] > 0.f) ? 1.f : 0.f;
      float w3 = (l3 == 1.f) ? 1.f : Lb / (Lb - pos3 + 1e-9f);
      lacc += w3 * bce_(p, l3);
    }
  }
  __syncthreads();
  if (lane == 0) sbuf[wave] = lacc;
  __syncthreads();
  if (tid == 0) {
    float s = 0.f;
    for (int w = 0; w < 8; ++w) s += sbuf[w];
    acc[8 + b] += s;
  }
}

__global__ void k_finalize(const float* __restrict__ acc, float* __restrict__ out) {
  int b = threadIdx.x;
  if (b < Bn) {
    out[b]      = acc[b] * 0.25f;                       // batch_spk_loss / C
    out[8 + b]  = acc[8 + b] / (acc[16 + b] + 1e-5f);   // fal / (valid_frames + 1e-5)
  }
}

// ---------------------------------------------------------------------------
// Host launcher
// ---------------------------------------------------------------------------
extern "C" void kernel_launch(void* const* d_in, const int* in_sizes, int n_in,
                              void* d_out, int out_size, void* d_ws, size_t ws_size,
                              hipStream_t stream) {
  (void)in_sizes; (void)n_in; (void)out_size; (void)ws_size;
  const float* enc   = (const float*)d_in[0];
  const int*   slen  = (const int*)  d_in[1];
  const int*   label = (const int*)  d_in[2];
  const float* Wp    = (const float*)d_in[3];
  const float* bp    = (const float*)d_in[4];
  const float* Wia   = (const float*)d_in[5];
  const float* Wha   = (const float*)d_in[6];
  const float* bia   = (const float*)d_in[7];
  const float* bha   = (const float*)d_in[8];
  const float* Wim   = (const float*)d_in[9];
  const float* Whm   = (const float*)d_in[10];
  const float* bim   = (const float*)d_in[11];
  const float* bhm   = (const float*)d_in[12];
  const float* h0    = (const float*)d_in[13];

  char* ws = (char*)d_ws;
  _Float16* enc_h  = (_Float16*)(ws + OFF_ENC_H);
  _Float16* wp_h   = (_Float16*)(ws + OFF_WP_H);
  _Float16* wia_h  = (_Float16*)(ws + OFF_WIA_H);
  _Float16* wha_h  = (_Float16*)(ws + OFF_WHA_H);
  _Float16* wim_h  = (_Float16*)(ws + OFF_WIM_H);
  _Float16* whm_h  = (_Float16*)(ws + OFF_WHM_H);
  float*    femb   = (float*)   (ws + OFF_FEMB);
  _Float16* femb_h = (_Float16*)(ws + OFF_FEMB_H);
  _Float16* seq_h  = (_Float16*)(ws + OFF_SEQ_H);
  _Float16* xph    = (_Float16*)(ws + OFF_XP);
  float*    outb   = (float*)   (ws + OFF_OUT);
  float*    updb   = (float*)   (ws + OFF_UPD);
  float*    hpm    = (float*)   (ws + OFF_HPM);
  float*    addh   = (float*)   (ws + OFF_ADDH);
  int*      perm   = (int*)     (ws + OFF_PERM);
  int*      mask   = (int*)     (ws + OFF_MASK);
  int*      act    = (int*)     (ws + OFF_ACT);
  int*      cexcl  = (int*)     (ws + OFF_CEXCL);
  float*    sif    = (float*)   (ws + OFF_SIF);
  int*      order  = (int*)     (ws + OFF_ORDER);
  float*    acc    = (float*)   (ws + OFF_ACC);

  k_init<<<1, 32, 0, stream>>>(acc);

  // f32 -> f16 staging for all GEMM operands
  k_f32_to_f16<<<4096, 256, 0, stream>>>(enc, enc_h, (int)(BTn * Dn));
  k_f32_to_f16<<<256, 256, 0, stream>>>(Wp,  wp_h,  Dn * Dn);
  k_f32_to_f16<<<256, 256, 0, stream>>>(Wia, wia_h, Gn * Dn);
  k_f32_to_f16<<<256, 256, 0, stream>>>(Wha, wha_h, Gn * Dn);
  k_f32_to_f16<<<256, 256, 0, stream>>>(Wim, wim_h, Gn * Dn);
  k_f32_to_f16<<<256, 256, 0, stream>>>(Whm, whm_h, Gn * Dn);

  // frame_emb = enc @ W_proj^T + b_proj   (WMMA; 2000x16 tiles, 8 waves/block)
  k_gemm_f16<<<4000, 256, 0, stream>>>(enc_h, wp_h, bp, femb, femb_h, (int)BTn, Dn);

  k_vad<<<Bn, 256, 0, stream>>>(label, slen, femb, h0, sif, acc);
  k_order<<<1, 64, 0, stream>>>(label, order);

  for (int k = 0; k < Cn; ++k) {
    k_perm<<<Bn, 32, 0, stream>>>(label, order, k, perm, mask, act, cexcl);
    k_gather_seq<<<8192, 256, 0, stream>>>(femb, perm, seq_h);
    // x_pre (adder) = seq @ Wi_a^T + bi_a   (f16 out; 2000x48 tiles)
    k_gemm_f16<<<12000, 256, 0, stream>>>(seq_h, wia_h, bia, (float*)nullptr, xph,
                                          (int)BTn, Gn);
    k_gru_scan<<<1, 512, 0, stream>>>(wha_h, bha, xph, mask, h0, outb, addh);
    k_score<<<Bn, 256, 0, stream>>>(femb, outb, act, cexcl, slen, acc);
    // x_pre (memory GRU), reusing the same f16 buffer
    k_gemm_f16<<<12000, 256, 0, stream>>>(seq_h, wim_h, bim, (float*)nullptr, xph,
                                          (int)BTn, Gn);
    k_hp_m<<<1, 512, 0, stream>>>(addh, whm_h, bhm, hpm);
    k_upd<<<8192, 256, 0, stream>>>(xph, hpm, addh, updb);
    k_scatter<<<Bn, 256, 0, stream>>>(updb, perm, mask, act, h0, sif, femb, femb_h, acc);
  }

  k_finalize<<<1, 32, 0, stream>>>(acc, (float*)d_out);
}
